// BBCUM_46686294508147
// MI455X (gfx1250) — 8-bit matrix operand order repaired and checked by reading, compiled, not run on hardware
//
#include <hip/hip_runtime.h>

typedef int   v8i  __attribute__((ext_vector_type(8)));
typedef int   v4i  __attribute__((ext_vector_type(4)));
typedef int   v2i  __attribute__((ext_vector_type(2)));
typedef float v8f  __attribute__((ext_vector_type(8)));

#define NF    64
#define KK    576
#define ASTRB 592

__device__ __forceinline__ unsigned char sgn_fp8(float v) {
    return (v > 0.f) ? 0x38u : ((v < 0.f) ? 0xB8u : 0x00u);
}
__device__ __forceinline__ float sgnf(float v) {
    return (v > 0.f) ? 1.f : ((v < 0.f) ? -1.f : 0.f);
}

__global__ void pack_bin_weights(const float* __restrict__ w,
                                 unsigned char* __restrict__ wp,
                                 float* __restrict__ scale, int rows) {
    __shared__ float red[64];
    int row = blockIdx.x;
    if (row >= rows) return;
    int t = threadIdx.x;
    const float* wr = w + (size_t)row * KK + t * 9;
    float s = 0.f;
#pragma unroll
    for (int i = 0; i < 9; ++i) {
        float v = wr[i];
        s += fabsf(v);
        wp[(size_t)row * KK + t * 9 + i] = sgn_fp8(v);
    }
    red[t] = s;
    __syncthreads();
    for (int off = 32; off > 0; off >>= 1) {
        if (t < off) red[t] += red[t + off];
        __syncthreads();
    }
    if (t == 0) scale[row] = red[0] / (float)KK;
}

__global__ void conv_first_kernel(const float* __restrict__ x,
                                  const float* __restrict__ w,
                                  const float* __restrict__ bias,
                                  float* __restrict__ out, int H, int W) {
    long i = (long)blockIdx.x * blockDim.x + threadIdx.x;
    long total = 4L * NF * H * W;
    if (i >= total) return;
    int wcol = (int)(i % W);
    long t = i / W;
    int h  = (int)(t % H); t /= H;
    int oc = (int)(t % NF);
    int b  = (int)(t / NF);
    float acc = bias[oc];
    for (int c = 0; c < 3; ++c)
        for (int r = 0; r < 3; ++r) {
            int hh = h + r - 1;
            if (hh < 0 || hh >= H) continue;
#pragma unroll
            for (int s = 0; s < 3; ++s) {
                int ww = wcol + s - 1;
                if (ww < 0 || ww >= W) continue;
                acc += x[((long)(b * 3 + c) * H + hh) * W + ww] *
                       w[((oc * 3 + c) * 3 + r) * 3 + s];
            }
        }
    out[i] = acc >= 0.f ? acc : 0.1f * acc;
}

template <int MODE, int CG, int PT>
__global__ void __launch_bounds__(CG* PT * 32)
bin_conv_gemm(const float* __restrict__ xin, const unsigned char* __restrict__ wp,
              const float* __restrict__ scale, const float* __restrict__ mv,
              const float* __restrict__ ap, const float* __restrict__ bp1,
              const float* __restrict__ bp2, float* __restrict__ out,
              int H, int W) {
    constexpr int TPW = PT * 16;
    __shared__ alignas(32) unsigned char Acol[TPW * ASTRB];
    const int wtile = blockIdx.x, h = blockIdx.y, b = blockIdx.z;
    const int tid = threadIdx.x;
    const int nthr = CG * PT * 32;
    const int wbase = wtile * TPW;

    for (int i = tid; i < TPW * KK; i += nthr) {
        int pix = i / KK;
        int k   = i - pix * KK;
        int c   = k / 9;
        int rs  = k - c * 9;
        int r3  = rs / 3;
        int r   = r3 - 1;
        int s   = rs - r3 * 3 - 1;
        int hh  = h + r;
        int ww  = wbase + pix + s;
        unsigned char v = 0u;
        if (hh >= 0 && hh < H && ww >= 0 && ww < W)
            v = sgn_fp8(xin[((long)(b * NF + c) * H + hh) * W + ww] + mv[c]);
        Acol[pix * ASTRB + k] = v;
    }
    __syncthreads();

    const int lane  = tid & 31;
    const int wv    = tid >> 5;
    const int pt    = wv % PT;
    const int cg    = wv / PT;
    const int half  = lane >> 4;
    const int nlane = lane & 15;
    const int ocb   = cg * 64;

    v8f acc0 = {0.f,0.f,0.f,0.f,0.f,0.f,0.f,0.f};
    v8f acc1 = acc0, acc2 = acc0, acc3 = acc0;
    union AU { v8i v; v2i d[4]; };
    union BU { v8i v; v4i q[2]; };

    const unsigned char* arow = &Acol[(pt * 16 + nlane) * ASTRB + half * 16];
    const unsigned char* wrow = wp + (size_t)(ocb + nlane) * KK + half * 16;

#pragma unroll 3
    for (int kc = 0; kc < 9; ++kc) {
        AU a;
        const unsigned char* ar = arow + kc * 64;
        a.d[0] = *(const v2i*)(ar);
        a.d[1] = *(const v2i*)(ar + 8);
        a.d[2] = *(const v2i*)(ar + 32);
        a.d[3] = *(const v2i*)(ar + 40);
        BU b0, b1, b2, b3;
        const unsigned char* wb = wrow + kc * 64;
        b0.q[0] = *(const v4i*)(wb);
        b0.q[1] = *(const v4i*)(wb + 32);
        b1.q[0] = *(const v4i*)(wb + 16 * KK);
        b1.q[1] = *(const v4i*)(wb + 16 * KK + 32);
        b2.q[0] = *(const v4i*)(wb + 32 * KK);
        b2.q[1] = *(const v4i*)(wb + 32 * KK + 32);
        b3.q[0] = *(const v4i*)(wb + 48 * KK);
        b3.q[1] = *(const v4i*)(wb + 48 * KK + 32);
        acc0 = __builtin_amdgcn_wmma_f32_16x16x64_fp8_fp8(a.v, b0.v, (short)0,
                                                          acc0, false, false);
        acc1 = __builtin_amdgcn_wmma_f32_16x16x64_fp8_fp8(a.v, b1.v, (short)0,
                                                          acc1, false, false);
        acc2 = __builtin_amdgcn_wmma_f32_16x16x64_fp8_fp8(a.v, b2.v, (short)0,
                                                          acc2, false, false);
        acc3 = __builtin_amdgcn_wmma_f32_16x16x64_fp8_fp8(a.v, b3.v, (short)0,
                                                          acc3, false, false);
    }

    auto epi = [&](v8f A, int nt) {
        int oc = ocb + nt * 16 + nlane;
        float sc = scale[oc], aa = ap[oc], B1 = bp1[oc], B2 = bp2[oc];
#pragma unroll
        for (int j = 0; j < 8; ++j) {
            int m   = j + 8 * half;
            int col = wbase + pt * 16 + m;
            float v = A[j] * sc + B1;
            v = (v >= 0.f) ? v : aa * v;
            v += B2;
            if (MODE == 0) {
                long idx = ((long)(b * NF + oc) * H + h) * W + col;
                out[idx] = v + xin[idx];
            } else {
                int cp = oc >> 2, r1 = (oc >> 1) & 1, r2 = oc & 1;
                float res = xin[((long)(b * NF + (oc & 63)) * H + h) * W + col];
                long oidx = ((long)(b * NF + cp) * (2L * H) + (2 * h + r1)) *
                                (2L * W) + (2 * col + r2);
                out[oidx] = v + res;
            }
        }
    };
    epi(acc0, 0); epi(acc1, 1); epi(acc2, 2); epi(acc3, 3);
}

__global__ void final_kernel(const float* __restrict__ hr,
                             const float* __restrict__ wl,
                             const float* __restrict__ bl,
                             const float* __restrict__ x0,
                             float* __restrict__ out,
                             int H2, int W2, int H0, int W0) {
    long i = (long)blockIdx.x * blockDim.x + threadIdx.x;
    long npix = 4L * H2 * W2;
    if (i >= npix) return;
    int wcol = (int)(i % W2);
    long t = i / W2;
    int h = (int)(t % H2);
    int b = (int)(t / H2);
    float acc[3] = {bl[0], bl[1], bl[2]};
    for (int c = 0; c < NF; ++c) {
#pragma unroll
        for (int r = 0; r < 3; ++r) {
            int hh = h + r - 1;
            if (hh < 0 || hh >= H2) continue;
#pragma unroll
            for (int s = 0; s < 3; ++s) {
                int ww = wcol + s - 1;
                if (ww < 0 || ww >= W2) continue;
                float v = hr[((long)(b * NF + c) * H2 + hh) * W2 + ww];
                v = (v >= 0.f) ? v : 0.1f * v;
#pragma unroll
                for (int oc = 0; oc < 3; ++oc)
                    acc[oc] += v * wl[((oc * NF + c) * 3 + r) * 3 + s];
            }
        }
    }
    float sy = (h + 0.5f) * 0.25f - 0.5f;
    float sx = (wcol + 0.5f) * 0.25f - 0.5f;
    int   y0 = (int)floorf(sy), xI = (int)floorf(sx);
    float fy = sy - y0, fx = sx - xI;
    int y0c = min(max(y0, 0), H0 - 1), y1c = min(max(y0 + 1, 0), H0 - 1);
    int x0c = min(max(xI, 0), W0 - 1), x1c = min(max(xI + 1, 0), W0 - 1);
#pragma unroll
    for (int oc = 0; oc < 3; ++oc) {
        const float* xp = x0 + ((long)(b * 3 + oc) * H0) * W0;
        float v00 = xp[(long)y0c * W0 + x0c], v01 = xp[(long)y0c * W0 + x1c];
        float v10 = xp[(long)y1c * W0 + x0c], v11 = xp[(long)y1c * W0 + x1c];
        float base = v00 * (1.f - fy) * (1.f - fx) + v01 * (1.f - fy) * fx +
                     v10 * fy * (1.f - fx)         + v11 * fy * fx;
        out[((long)(b * 3 + oc) * H2 + h) * W2 + wcol] = acc[oc] + base;
    }
}

extern "C" void kernel_launch(void* const* d_in, const int* in_sizes, int n_in,
                              void* d_out, int out_size, void* d_ws, size_t ws_size,
                              hipStream_t stream) {
    (void)in_sizes; (void)n_in; (void)out_size; (void)ws_size;
    const float* x         = (const float*)d_in[0];
    const float* cfw       = (const float*)d_in[1];
    const float* cfb       = (const float*)d_in[2];
    const float* body_move = (const float*)d_in[3];
    const float* body_w    = (const float*)d_in[4];
    const float* body_a    = (const float*)d_in[5];
    const float* body_b1   = (const float*)d_in[6];
    const float* body_b2   = (const float*)d_in[7];
    const float* up1_move  = (const float*)d_in[8];
    const float* up1_w     = (const float*)d_in[9];
    const float* up1_a     = (const float*)d_in[10];
    const float* up1_b1    = (const float*)d_in[11];
    const float* up1_b2    = (const float*)d_in[12];
    const float* up2_move  = (const float*)d_in[13];
    const float* up2_w     = (const float*)d_in[14];
    const float* up2_a     = (const float*)d_in[15];
    const float* up2_b1    = (const float*)d_in[16];
    const float* up2_b2    = (const float*)d_in[17];
    const float* hr_move   = (const float*)d_in[18];
    const float* hr_w      = (const float*)d_in[19];
    const float* hr_a      = (const float*)d_in[20];
    const float* hr_b1     = (const float*)d_in[21];
    const float* hr_b2     = (const float*)d_in[22];
    const float* clw       = (const float*)d_in[23];
    const float* clb       = (const float*)d_in[24];
    float* out = (float*)d_out;

    char* ws = (char*)d_ws;
    size_t off = 0;
    auto alloc = [&](size_t bytes) {
        void* p = ws + off;
        off = (off + bytes + 255) & ~(size_t)255;
        return p;
    };
    unsigned char* wp_body = (unsigned char*)alloc(2048UL * KK);
    unsigned char* wp_up1  = (unsigned char*)alloc(256UL * KK);
    unsigned char* wp_up2  = (unsigned char*)alloc(256UL * KK);
    unsigned char* wp_hr   = (unsigned char*)alloc(64UL * KK);
    float* sc_body = (float*)alloc(2048UL * sizeof(float));
    float* sc_up1  = (float*)alloc(256UL * sizeof(float));
    float* sc_up2  = (float*)alloc(256UL * sizeof(float));
    float* sc_hr   = (float*)alloc(64UL * sizeof(float));
    float* featA   = (float*)alloc(4UL * NF * 128 * 128 * sizeof(float));
    float* featB   = (float*)alloc(4UL * NF * 128 * 128 * sizeof(float));
    float* up1o    = (float*)alloc(4UL * NF * 256 * 256 * sizeof(float));
    float* up2o    = (float*)alloc(4UL * NF * 512 * 512 * sizeof(float));
    float* hro     = (float*)alloc(4UL * NF * 512 * 512 * sizeof(float));

    pack_bin_weights<<<2048, 64, 0, stream>>>(body_w, wp_body, sc_body, 2048);
    pack_bin_weights<<<256,  64, 0, stream>>>(up1_w,  wp_up1,  sc_up1,  256);
    pack_bin_weights<<<256,  64, 0, stream>>>(up2_w,  wp_up2,  sc_up2,  256);
    pack_bin_weights<<<64,   64, 0, stream>>>(hr_w,   wp_hr,   sc_hr,   64);

    {
        long n = 4L * NF * 128 * 128;
        conv_first_kernel<<<(unsigned)((n + 255) / 256), 256, 0, stream>>>(
            x, cfw, cfb, featA, 128, 128);
    }

    float* cur = featA; float* nxt = featB;
    for (int u = 0; u < 32; ++u) {
        dim3 g(128 / 64, 128, 4);
        bin_conv_gemm<0, 1, 4><<<g, 128, 0, stream>>>(
            cur, wp_body + (size_t)u * 64 * KK, sc_body + u * 64,
            body_move + u * 64, body_a + u * 64, body_b1 + u * 64,
            body_b2 + u * 64, nxt, 128, 128);
        float* tmp = cur; cur = nxt; nxt = tmp;
    }

    {   dim3 g(128 / 32, 128, 4);
        bin_conv_gemm<1, 4, 2><<<g, 256, 0, stream>>>(
            cur, wp_up1, sc_up1, up1_move, up1_a, up1_b1, up1_b2, up1o, 128, 128);
    }
    {   dim3 g(256 / 32, 256, 4);
        bin_conv_gemm<1, 4, 2><<<g, 256, 0, stream>>>(
            up1o, wp_up2, sc_up2, up2_move, up2_a, up2_b1, up2_b2, up2o, 256, 256);
    }
    {   dim3 g(512 / 64, 512, 4);
        bin_conv_gemm<0, 1, 4><<<g, 128, 0, stream>>>(
            up2o, wp_hr, sc_hr, hr_move, hr_a, hr_b1, hr_b2, hro, 512, 512);
    }
    {
        long n = 4L * 512 * 512;
        final_kernel<<<(unsigned)((n + 127) / 128), 128, 0, stream>>>(
            hro, clw, clb, x, out, 512, 512, 128, 128);
    }
}
